// SimpleGNN_66116726555132
// MI455X (gfx1250) — matrix operand order repaired and checked by reading, not compiled, not run on hardware
//
#include <hip/hip_runtime.h>
#include <hip/hip_bf16.h>

// ---------------------------------------------------------------------------
// Types for WMMA fragments (wave32, gfx1250)
// ---------------------------------------------------------------------------
typedef __attribute__((ext_vector_type(16))) _Float16 v16h;
typedef __attribute__((ext_vector_type(8)))  _Float16 v8h;
typedef __attribute__((ext_vector_type(4)))  _Float16 v4h;
typedef __attribute__((ext_vector_type(8)))  float    v8f;

#define G_GRAPHS 64

__device__ __forceinline__ v8f wmma_f16(v16h a, v16h b, v8f c) {
  // D = A(16x32) * B(32x16) + C, f32 accumulate
  return __builtin_amdgcn_wmma_f32_16x16x32_f16(
      /*neg_a=*/false, a, /*neg_b=*/false, b,
      /*c_mod=*/(short)0, c, /*reuse_a=*/false, /*reuse_b=*/false);
}

// A fragment: 16x32 f16 tile, rows rowBase..rowBase+15 from LDS [.. , ld]
__device__ __forceinline__ v16h ldsA(const _Float16* base, int ld, int rowBase,
                                     int k0, int lane) {
  int l16 = lane & 15, grp = lane >> 4;
  const _Float16* p = base + (rowBase + l16) * ld + k0 + grp * 8;
  v8h lo = *(const v8h*)p;
  v8h hi = *(const v8h*)(p + 16);
  return __builtin_shufflevector(lo, hi, 0, 1, 2, 3, 4, 5, 6, 7,
                                 8, 9, 10, 11, 12, 13, 14, 15);
}

// B fragment: 32x16 f16 tile from LDS weights stored TRANSPOSED: bt[n*ld + k]
__device__ __forceinline__ v16h ldsB(const _Float16* bt, int ld, int colBase,
                                     int k0, int lane) {
  int l16 = lane & 15, grp = lane >> 4;
  const _Float16* p = bt + (colBase + l16) * ld + k0 + grp * 8;
  v8h lo = *(const v8h*)p;
  v8h hi = *(const v8h*)(p + 16);
  return __builtin_shufflevector(lo, hi, 0, 1, 2, 3, 4, 5, 6, 7,
                                 8, 9, 10, 11, 12, 13, 14, 15);
}

// ---------------------------------------------------------------------------
// CDNA5 async global -> LDS copy (CDNA5 ISA).
// LDS dest offset comes from truncating the generic pointer (LDS aperture:
// low 32 bits of the address); tracked by ASYNCcnt, drained with s_wait_asynccnt.
// ---------------------------------------------------------------------------
__device__ __forceinline__ unsigned lds_off(const void* p) {
  return (unsigned)(unsigned long long)p;
}
__device__ __forceinline__ void async_ld_b128(unsigned ldst, const void* gsrc) {
  asm volatile("global_load_async_to_lds_b128 %0, %1, off"
               :
               : "v"(ldst), "v"(gsrc)
               : "memory");
}
__device__ __forceinline__ void wait_async0() {
  asm volatile("s_wait_asynccnt 0" ::: "memory");
}

// ---------------------------------------------------------------------------
// Fused 3-layer MLP:  out = L3( prelu(L2( prelu(L1(in)) )) )
// 128-row tile per block, 8 waves, each wave owns 16 rows.
// IN_MODE 0: f32 input A[M,K0]
// IN_MODE 1: f16 input Ah[M,K0], async-copied global->LDS
// IN_MODE 2: f16 gather, row m = concat(Ah[er[m]][0:64], Ah[ec[m]][0:64]),
//            async-copied global->LDS (K0 must be 128)
// ---------------------------------------------------------------------------
template <int K0, int N1, int N2, int N3, bool BIAS, int IN_MODE>
__global__ __launch_bounds__(256) void fused_mlp3_kernel(
    const float* __restrict__ A, const _Float16* __restrict__ Ah,
    const int* __restrict__ er, const int* __restrict__ ec,
    const float* __restrict__ W1, const float* __restrict__ b1,
    const float* __restrict__ pa1,
    const float* __restrict__ W2, const float* __restrict__ b2,
    const float* __restrict__ pa2,
    const float* __restrict__ W3, const float* __restrict__ b3,
    float* __restrict__ out, int M) {
  constexpr int K0P = (K0 + 31) & ~31;          // pad K to WMMA K=32
  constexpr int N3P = (N3 < 16) ? 16 : N3;      // pad last layer to tile width
  constexpr int NA  = (N1 > N2) ? N1 : N2;      // activation buffer width

  __shared__ alignas(16) _Float16 sIn[128 * K0P];
  __shared__ alignas(16) _Float16 sW1[N1 * K0P];   // transposed [n][k]
  __shared__ alignas(16) _Float16 sW2[N2 * N1];
  __shared__ alignas(16) _Float16 sW3[N3P * N2];
  __shared__ alignas(16) _Float16 sAct[128 * NA];
  __shared__ float sB1[N1], sB2[N2], sB3[N3P];

  const int tid = threadIdx.x;
  const int m0  = blockIdx.x * 128;

  // ---- stage input tile first (async path overlaps with weight staging) ----
  if constexpr (IN_MODE == 1) {
    constexpr int CPR = K0 / 8;                 // 16B chunks per row
    unsigned lbase = lds_off(sIn);
    for (int idx = tid; idx < 128 * CPR; idx += 256) {
      int r = idx / CPR, c8 = idx - r * CPR;
      int m = m0 + r; if (m >= M) m = M - 1;
      async_ld_b128(lbase + (unsigned)(r * K0 + c8 * 8) * 2,
                    (const char*)Ah + ((long)m * K0 + c8 * 8) * 2);
    }
  } else if constexpr (IN_MODE == 2) {
    // per row: two 64-f16 segments (row node, col node), 8 x 16B chunks each
    unsigned lbase = lds_off(sIn);
    for (int idx = tid; idx < 128 * 2 * 8; idx += 256) {
      int seg = idx >> 3, part = idx & 7;
      int r = seg >> 1, half = seg & 1;
      int m = m0 + r; if (m >= M) m = M - 1;
      int node = half ? ec[m] : er[m];
      async_ld_b128(lbase + (unsigned)(r * 128 + half * 64 + part * 8) * 2,
                    (const char*)Ah + ((long)node * 64 + part * 8) * 2);
    }
  }

  // ---- stage weights (coalesced global reads, transposed LDS stores) ----
  for (int idx = tid; idx < K0 * N1; idx += 256) {
    int k = idx / N1, n = idx - k * N1;
    sW1[n * K0P + k] = (_Float16)W1[idx];
  }
  if constexpr (K0P > K0) {
    for (int idx = tid; idx < (K0P - K0) * N1; idx += 256) {
      int kk = idx / N1, n = idx - kk * N1;
      sW1[n * K0P + K0 + kk] = (_Float16)0.f;
    }
  }
  for (int idx = tid; idx < N1 * N2; idx += 256) {
    int k = idx / N2, n = idx - k * N2;
    sW2[n * N1 + k] = (_Float16)W2[idx];
  }
  for (int idx = tid; idx < N2 * N3; idx += 256) {
    int k = idx / N3, n = idx - k * N3;
    sW3[n * N2 + k] = (_Float16)W3[idx];
  }
  if constexpr (N3P > N3) {
    for (int idx = tid; idx < (N3P - N3) * N2; idx += 256) {
      int nn = idx / N2, k = idx - nn * N2;
      sW3[(N3 + nn) * N2 + k] = (_Float16)0.f;
    }
  }
  if (BIAS) {
    for (int n = tid; n < N1; n += 256) sB1[n] = b1[n];
    for (int n = tid; n < N2; n += 256) sB2[n] = b2[n];
    for (int n = tid; n < N3P; n += 256) sB3[n] = (n < N3) ? b3[n] : 0.f;
  }

  // ---- f32 input tile staging (vectorized) ----
  if constexpr (IN_MODE == 0) {
    constexpr int KC = K0 / 4;
    for (int idx = tid; idx < 128 * KC; idx += 256) {
      int r = idx / KC, k4 = idx - r * KC;
      int m = m0 + r;
      float4 v = make_float4(0.f, 0.f, 0.f, 0.f);
      if (m < M) v = ((const float4*)A)[(long)m * KC + k4];
      v4h h4 = {(_Float16)v.x, (_Float16)v.y, (_Float16)v.z, (_Float16)v.w};
      *(v4h*)(sIn + r * K0P + k4 * 4) = h4;
    }
    if constexpr (K0P > K0) {
      constexpr int ZC = (K0P - K0) / 4;
      for (int idx = tid; idx < 128 * ZC; idx += 256) {
        int r = idx / ZC, k4 = idx - r * ZC;
        v4h z = {(_Float16)0.f, (_Float16)0.f, (_Float16)0.f, (_Float16)0.f};
        *(v4h*)(sIn + r * K0P + K0 + k4 * 4) = z;
      }
    }
  } else {
    wait_async0();   // drain ASYNCcnt before the barrier publishes sIn
  }
  __syncthreads();

  const int wave = tid >> 5, lane = tid & 31;
  const int rowBase = wave * 16;
  const int l16 = lane & 15, grp = lane >> 4;
  const float a1v = pa1[0];
  const float a2v = pa2[0];

  // ---- layer 1: sIn -> sAct ----
  #pragma unroll
  for (int ct = 0; ct < N1 / 16; ++ct) {
    v8f acc = {};
    #pragma unroll
    for (int k0 = 0; k0 < K0P; k0 += 32)
      acc = wmma_f16(ldsA(sIn, K0P, rowBase, k0, lane),
                     ldsB(sW1, K0P, ct * 16, k0, lane), acc);
    int colg = ct * 16 + l16;
    float bb = BIAS ? sB1[colg] : 0.f;
    #pragma unroll
    for (int v = 0; v < 8; ++v) {
      float xv = acc[v] + bb;
      xv = (xv >= 0.f) ? xv : a1v * xv;
      sAct[(rowBase + grp * 8 + v) * NA + colg] = (_Float16)xv;
    }
  }
  // ---- layer 2: sAct -> regs -> sAct (all reads before overwrite) ----
  v8f acc2[N2 / 16];
  #pragma unroll
  for (int ct = 0; ct < N2 / 16; ++ct) {
    v8f acc = {};
    #pragma unroll
    for (int k0 = 0; k0 < N1; k0 += 32)
      acc = wmma_f16(ldsA(sAct, NA, rowBase, k0, lane),
                     ldsB(sW2, N1, ct * 16, k0, lane), acc);
    acc2[ct] = acc;
  }
  #pragma unroll
  for (int ct = 0; ct < N2 / 16; ++ct) {
    int colg = ct * 16 + l16;
    float bb = BIAS ? sB2[colg] : 0.f;
    #pragma unroll
    for (int v = 0; v < 8; ++v) {
      float xv = acc2[ct][v] + bb;
      xv = (xv >= 0.f) ? xv : a2v * xv;
      sAct[(rowBase + grp * 8 + v) * NA + colg] = (_Float16)xv;
    }
  }
  // ---- layer 3: sAct -> out ----
  #pragma unroll
  for (int ct = 0; ct < N3P / 16; ++ct) {
    v8f acc = {};
    #pragma unroll
    for (int k0 = 0; k0 < N2; k0 += 32)
      acc = wmma_f16(ldsA(sAct, NA, rowBase, k0, lane),
                     ldsB(sW3, N2, ct * 16, k0, lane), acc);
    int colg = ct * 16 + l16;
    if (colg < N3) {
      float bb = BIAS ? sB3[colg] : 0.f;
      #pragma unroll
      for (int v = 0; v < 8; ++v) {
        int m = m0 + rowBase + grp * 8 + v;
        if (m < M) out[(long)m * N3 + colg] = acc[v] + bb;
      }
    }
  }
}

// ---------------------------------------------------------------------------
// Plain GEMM: out[M,NOUT] = A[M,K] @ W[K,NOUT]   (f32 in/out, no bias)
// ---------------------------------------------------------------------------
template <int K, int NOUT>
__global__ __launch_bounds__(256) void gemm_kernel(
    const float* __restrict__ A, const float* __restrict__ W,
    float* __restrict__ out, int M) {
  __shared__ alignas(16) _Float16 sIn[128 * K];
  __shared__ alignas(16) _Float16 sW[NOUT * K];  // transposed [n][k]

  const int tid = threadIdx.x;
  const int m0  = blockIdx.x * 128;

  for (int idx = tid; idx < K * NOUT; idx += 256) {
    int k = idx / NOUT, n = idx - k * NOUT;
    sW[n * K + k] = (_Float16)W[idx];
  }
  constexpr int KC = K / 4;
  for (int idx = tid; idx < 128 * KC; idx += 256) {
    int r = idx / KC, k4 = idx - r * KC;
    int m = m0 + r;
    float4 v = make_float4(0.f, 0.f, 0.f, 0.f);
    if (m < M) v = ((const float4*)A)[(long)m * KC + k4];
    v4h h4 = {(_Float16)v.x, (_Float16)v.y, (_Float16)v.z, (_Float16)v.w};
    *(v4h*)(sIn + r * K + k4 * 4) = h4;
  }
  __syncthreads();

  const int wave = tid >> 5, lane = tid & 31;
  const int rowBase = wave * 16;
  const int l16 = lane & 15, grp = lane >> 4;

  #pragma unroll
  for (int ct = 0; ct < NOUT / 16; ++ct) {
    v8f acc = {};
    #pragma unroll
    for (int k0 = 0; k0 < K; k0 += 32)
      acc = wmma_f16(ldsA(sIn, K, rowBase, k0, lane),
                     ldsB(sW, K, ct * 16, k0, lane), acc);
    int colg = ct * 16 + l16;
    #pragma unroll
    for (int v = 0; v < 8; ++v) {
      int m = m0 + rowBase + grp * 8 + v;
      if (m < M) out[(long)m * NOUT + colg] = acc[v];
    }
  }
}

// ---------------------------------------------------------------------------
// Elementwise / scatter helpers
// ---------------------------------------------------------------------------
__global__ void zero_kernel(float* __restrict__ p, long n) {
  long i = (long)blockIdx.x * 256 + threadIdx.x;
  if (i < n) p[i] = 0.f;
}

__global__ void pool_cnt_kernel(const int* __restrict__ batch,
                                float* __restrict__ cnt, int N) {
  int i = blockIdx.x * 256 + threadIdx.x;
  if (i < N) atomicAdd(&cnt[batch[i]], 1.f);
}

__global__ void pool_scatter_kernel(const float* __restrict__ xe,
                                    const int* __restrict__ batch,
                                    float* __restrict__ g, int N) {
  long i = (long)blockIdx.x * 256 + threadIdx.x;
  if (i >= (long)N * 64) return;
  int n = (int)(i >> 6), c = (int)(i & 63);
  atomicAdd(&g[batch[n] * 64 + c], xe[i]);
}

__global__ void pool_div_kernel(float* __restrict__ g,
                                const float* __restrict__ cnt) {
  int i = blockIdx.x * 256 + threadIdx.x;
  if (i < G_GRAPHS * 64) g[i] /= fmaxf(cnt[i >> 6], 1.f);
}

__global__ void build_h_kernel(const float* __restrict__ xe,
                               const float* __restrict__ ge,
                               const int* __restrict__ batch,
                               float* __restrict__ h, int N) {
  long i = (long)blockIdx.x * 256 + threadIdx.x;
  if (i >= (long)N * 128) return;
  int n = (int)(i >> 7), c = (int)(i & 127);
  h[i] = (c < 64) ? xe[(long)n * 64 + c] : ge[batch[n] * 64 + (c - 64)];
}

__global__ void deg_init_kernel(float* __restrict__ deg, int N) {
  int i = blockIdx.x * 256 + threadIdx.x;
  if (i < N) deg[i] = 1.f;  // self loop
}

__global__ void deg_count_kernel(const int* __restrict__ col,
                                 float* __restrict__ deg, int E) {
  int i = blockIdx.x * 256 + threadIdx.x;
  if (i < E) atomicAdd(&deg[col[i]], 1.f);
}

__global__ void dinv_kernel(float* __restrict__ deg, int N) {
  int i = blockIdx.x * 256 + threadIdx.x;
  if (i < N) { float d = deg[i]; deg[i] = (d > 0.f) ? rsqrtf(d) : 0.f; }
}

// one wave per edge; lanes sweep channels -> coalesced loads + atomics
template <int C>
__global__ void gcn_scatter_kernel(const int* __restrict__ row,
                                   const int* __restrict__ col,
                                   const float* __restrict__ dinv,
                                   const float* __restrict__ xw,
                                   float* __restrict__ t, int E) {
  int w = (blockIdx.x * 256 + threadIdx.x) >> 5;
  int lane = threadIdx.x & 31;
  if (w >= E) return;
  int r = row[w], c = col[w];
  float nrm = dinv[r] * dinv[c];
  #pragma unroll
  for (int i = 0; i < C / 32; ++i) {
    int ch = lane + 32 * i;
    atomicAdd(&t[(long)c * C + ch], xw[(long)r * C + ch] * nrm);
  }
}

// layer1: h += relu(t + self + b)
__global__ void gcn_combine1_kernel(const float* __restrict__ t,
                                    const float* __restrict__ xw,
                                    const float* __restrict__ dinv,
                                    const float* __restrict__ b,
                                    float* __restrict__ h, int N) {
  long i = (long)blockIdx.x * 256 + threadIdx.x;
  if (i >= (long)N * 128) return;
  int n = (int)(i >> 7), c = (int)(i & 127);
  float di = dinv[n];
  float v = t[i] + xw[i] * di * di + b[c];
  h[i] += fmaxf(v, 0.f);
}

// layer2 + residual: fe = xe + (t + self + b), stored as f16 for the MLPs
__global__ void gcn_combine2_fe_kernel(const float* __restrict__ t,
                                       const float* __restrict__ xw,
                                       const float* __restrict__ dinv,
                                       const float* __restrict__ b,
                                       const float* __restrict__ xe,
                                       _Float16* __restrict__ feh, int N) {
  long i = (long)blockIdx.x * 256 + threadIdx.x;
  if (i >= (long)N * 64) return;
  int n = (int)(i >> 6), c = (int)(i & 63);
  float di = dinv[n];
  feh[i] = (_Float16)(xe[i] + (t[i] + xw[i] * di * di + b[c]));
}

// ---------------------------------------------------------------------------
// Host side
// ---------------------------------------------------------------------------
struct Mlp {
  const float *W0, *W1, *W2, *b0, *b1, *b2, *a0, *a1;
};

static int parse_mlp(void* const* din, const int* sz, int i, bool bias, Mlp& m) {
  m.W0 = (const float*)din[i++];
  m.W1 = (const float*)din[i++];
  m.W2 = (const float*)din[i++];
  if (bias) {
    if (sz[i] == 1) {  // sorted dict order: W, a, b
      m.a0 = (const float*)din[i++]; m.a1 = (const float*)din[i++];
      m.b0 = (const float*)din[i++]; m.b1 = (const float*)din[i++];
      m.b2 = (const float*)din[i++];
    } else {           // insertion order: W, b, a
      m.b0 = (const float*)din[i++]; m.b1 = (const float*)din[i++];
      m.b2 = (const float*)din[i++];
      m.a0 = (const float*)din[i++]; m.a1 = (const float*)din[i++];
    }
  } else {             // bias=None leaves dropped from pytree
    m.b0 = m.b1 = m.b2 = nullptr;
    m.a0 = (const float*)din[i++]; m.a1 = (const float*)din[i++];
  }
  return i;
}

static inline int cdiv(long a, long b) { return (int)((a + b - 1) / b); }

extern "C" void kernel_launch(void* const* d_in, const int* in_sizes, int n_in,
                              void* d_out, int out_size, void* d_ws,
                              size_t ws_size, hipStream_t stream) {
  int i = 0;
  const float* x = (const float*)d_in[i++];
  const int N = in_sizes[0] / 16;

  Mlp embed; i = parse_mlp(d_in, in_sizes, i, true, embed);
  Mlp genc;  i = parse_mlp(d_in, in_sizes, i, true, genc);
  const float* cW1 = (const float*)d_in[i++];
  const float* cb1 = (const float*)d_in[i++];
  const float* cW2 = (const float*)d_in[i++];
  const float* cb2 = (const float*)d_in[i++];
  Mlp nodep; i = parse_mlp(d_in, in_sizes, i, false, nodep);
  Mlp edgep; i = parse_mlp(d_in, in_sizes, i, true, edgep);
  const int* eidx = (const int*)d_in[i];
  const int E = in_sizes[i] / 2; i++;
  const int* batch = (const int*)d_in[i++];
  const int* erow = eidx;
  const int* ecol = eidx + E;

  // workspace layout (floats)
  float* w   = (float*)d_ws;
  float* xe  = w;                  // [N,64]
  float* h   = xe + (long)N * 64;  // [N,128]
  float* xw  = h + (long)N * 128;  // [N,128] (reused for layer2 [N,64])
  float* t   = xw + (long)N * 128; // [N,128] scatter target (reused)
  float* deg = t + (long)N * 128;  // [N]     (becomes dinv)
  float* g   = deg + N;            // [G,64]
  float* ge  = g + G_GRAPHS * 64;  // [G,64]
  float* cnt = ge + G_GRAPHS * 64; // [G]
  _Float16* feh = (_Float16*)(((unsigned long long)(cnt + G_GRAPHS) + 15ULL)
                              & ~15ULL);       // [N,64] f16 final features

  float* node_out = (float*)d_out;             // [N,4]
  float* edge_out = node_out + (long)N * 4;    // [E,4]

  const int nb = cdiv(N, 128);
  const int eb = cdiv(E, 128);

  // 1. embed MLP: x[N,16] -> xe[N,64]
  fused_mlp3_kernel<16, 128, 128, 64, true, 0><<<nb, 256, 0, stream>>>(
      x, nullptr, nullptr, nullptr,
      embed.W0, embed.b0, embed.a0, embed.W1, embed.b1, embed.a1,
      embed.W2, embed.b2, xe, N);

  // 2. global mean pool
  zero_kernel<<<cdiv(G_GRAPHS * 64, 256), 256, 0, stream>>>(g, G_GRAPHS * 64);
  zero_kernel<<<1, 256, 0, stream>>>(cnt, G_GRAPHS);
  pool_cnt_kernel<<<cdiv(N, 256), 256, 0, stream>>>(batch, cnt, N);
  pool_scatter_kernel<<<cdiv((long)N * 64, 256), 256, 0, stream>>>(xe, batch, g, N);
  pool_div_kernel<<<cdiv(G_GRAPHS * 64, 256), 256, 0, stream>>>(g, cnt);

  // 3. graph encoder MLP: g[G,64] -> ge[G,64]
  fused_mlp3_kernel<64, 128, 128, 64, true, 0><<<1, 256, 0, stream>>>(
      g, nullptr, nullptr, nullptr,
      genc.W0, genc.b0, genc.a0, genc.W1, genc.b1, genc.a1,
      genc.W2, genc.b2, ge, G_GRAPHS);

  // 4. h = concat(xe, ge[batch])
  build_h_kernel<<<cdiv((long)N * 128, 256), 256, 0, stream>>>(xe, ge, batch, h, N);

  // 5. degrees (cols + self loops) -> dinv
  deg_init_kernel<<<cdiv(N, 256), 256, 0, stream>>>(deg, N);
  deg_count_kernel<<<cdiv(E, 256), 256, 0, stream>>>(ecol, deg, E);
  dinv_kernel<<<cdiv(N, 256), 256, 0, stream>>>(deg, N);

  // 6. GCN layer 1: xw = h @ W1; t = scatter; h += relu(t + self + b1)
  gemm_kernel<128, 128><<<nb, 256, 0, stream>>>(h, cW1, xw, N);
  zero_kernel<<<cdiv((long)N * 128, 256), 256, 0, stream>>>(t, (long)N * 128);
  gcn_scatter_kernel<128><<<cdiv((long)E * 32, 256), 256, 0, stream>>>(
      erow, ecol, deg, xw, t, E);
  gcn_combine1_kernel<<<cdiv((long)N * 128, 256), 256, 0, stream>>>(
      t, xw, deg, cb1, h, N);

  // 7. GCN layer 2 + residual: feh = f16(xe + scatter(h@W2) + self + b2)
  gemm_kernel<128, 64><<<nb, 256, 0, stream>>>(h, cW2, xw, N);
  zero_kernel<<<cdiv((long)N * 64, 256), 256, 0, stream>>>(t, (long)N * 64);
  gcn_scatter_kernel<64><<<cdiv((long)E * 32, 256), 256, 0, stream>>>(
      erow, ecol, deg, xw, t, E);
  gcn_combine2_fe_kernel<<<cdiv((long)N * 64, 256), 256, 0, stream>>>(
      t, xw, deg, cb2, xe, feh, N);

  // 8. node predictions: feh[N,64] -> [N,4]  (no bias; async f16 staging)
  fused_mlp3_kernel<64, 64, 64, 4, false, 1><<<nb, 256, 0, stream>>>(
      nullptr, feh, nullptr, nullptr,
      nodep.W0, nullptr, nodep.a0, nodep.W1, nullptr, nodep.a1,
      nodep.W2, nullptr, node_out, N);

  // 9. edge predictions: concat(feh[row], feh[col])[E,128] -> [E,4]
  //    (async gathered f16 staging)
  fused_mlp3_kernel<128, 64, 64, 4, true, 2><<<eb, 256, 0, stream>>>(
      nullptr, feh, erow, ecol,
      edgep.W0, edgep.b0, edgep.a0, edgep.W1, edgep.b1, edgep.a1,
      edgep.W2, edgep.b2, edge_out, E);
}
